// MPConv_21483426414655
// MI455X (gfx1250) — compile-verified, matrix operand order checked by reading, not run on hardware
//
#include <hip/hip_runtime.h>

typedef _Float16 half_t;
typedef __attribute__((ext_vector_type(16))) _Float16 v16h;
typedef __attribute__((ext_vector_type(8)))  _Float16 v8h;
typedef __attribute__((ext_vector_type(8)))  float    v8f;
typedef __attribute__((ext_vector_type(4)))  float    v4f;

#define N_NODES_C   100000
#define N_EDGES_C   1600000
#define OUT_CH_C    64
#define NEG_SLOPE_C 0.01f

// ---- packed weight (f16, WMMA A-fragment layout of W^T) offsets, halves ----
#define FRAG_HALVES 512                       // 32 lanes * 16 halves
#define L0_FRAGS    20                        // 5 kchunks * 4 ftiles
#define HID_FRAGS   8                         // 2 kchunks * 4 ftiles
#define L0_OFF      0
#define HID_OFF     (L0_FRAGS * FRAG_HALVES)                // 10240
#define OUT_OFF     (HID_OFF + 4 * HID_FRAGS * FRAG_HALVES) // 26624
#define WPK_HALVES  (OUT_OFF + HID_FRAGS * FRAG_HALVES)     // 30720

#define WAVES_PER_BLOCK 8
#define EDGES_PER_WAVE  32                    // two 16-edge column groups
#define BLOCK_THREADS   (WAVES_PER_BLOCK * 32)
#define EDGES_PER_BLOCK (WAVES_PER_BLOCK * EDGES_PER_WAVE)  // 256

// ---------------------------------------------------------------------------
// Pack weights from reference [K][N] f32 layout into f16 fragment layout.
// Fragment (kchunk, ftile): lane owns W^T row m = ftile*16 + (lane&15)
// (i.e. output feature); half j holds k_local = kbase + j + (j<8 ? 0 : 8),
// kbase = (lane>>4)*8.  (ISA 16-bit 16x32 A layout.)
// ---------------------------------------------------------------------------
__global__ void pack_weights_kernel(const float* __restrict__ W_in,
                                    const float* __restrict__ W_h,
                                    const float* __restrict__ W_out,
                                    half_t* __restrict__ wpk) {
    int idx = blockIdx.x * blockDim.x + threadIdx.x;
    if (idx >= WPK_HALVES) return;

    const float* W;
    int frag, rel;
    if (idx < HID_OFF) {                       // input layer: K = 160
        frag = idx / FRAG_HALVES;
        rel  = idx % FRAG_HALVES;
        W    = W_in;
    } else if (idx < OUT_OFF) {                // 4 hidden layers: K = 64
        int r     = idx - HID_OFF;
        int layer = r / (HID_FRAGS * FRAG_HALVES);
        int rr    = r % (HID_FRAGS * FRAG_HALVES);
        frag = rr / FRAG_HALVES;
        rel  = rr % FRAG_HALVES;
        W    = W_h + (size_t)layer * OUT_CH_C * OUT_CH_C;
    } else {                                   // output layer: K = 64
        int r = idx - OUT_OFF;
        frag = r / FRAG_HALVES;
        rel  = r % FRAG_HALVES;
        W    = W_out;
    }
    int lane   = rel / 16;
    int j      = rel % 16;
    int kchunk = frag / 4;
    int ftile  = frag % 4;
    int fo     = ftile * 16 + (lane & 15);     // output feature (row of W^T)
    int kbase  = (lane >> 4) * 8;
    int klocal = kbase + j + ((j < 8) ? 0 : 8);
    int k      = kchunk * 32 + klocal;         // input feature
    wpk[idx] = (half_t)W[(size_t)k * OUT_CH_C + fo];
}

__global__ void zero_kernel(float* __restrict__ p, int n) {
    int i = blockIdx.x * blockDim.x + threadIdx.x;
    if (i < n) p[i] = 0.0f;
}

// ---------------------------------------------------------------------------
// helpers
// ---------------------------------------------------------------------------
__device__ __forceinline__ v8f vzero8() {
    v8f z = {0.f, 0.f, 0.f, 0.f, 0.f, 0.f, 0.f, 0.f};
    return z;
}

__device__ __forceinline__ v16h cat8(v8h lo, v8h hi) {
    return __builtin_shufflevector(lo, hi, 0, 1, 2, 3, 4, 5, 6, 7,
                                           8, 9, 10, 11, 12, 13, 14, 15);
}

// Weight (A-operand) fragment from LDS: 2x ds_load_b128 per lane.
__device__ __forceinline__ v16h load_w_lds(const half_t* lds_w,
                                           int frag_off_halves, int lane) {
    const half_t* p = lds_w + frag_off_halves + lane * 16;
    v8h lo = *(const v8h*)(p);
    v8h hi = *(const v8h*)(p + 8);
    return cat8(lo, hi);
}

// Layer-0 B fragment (edge-feature columns) gathered straight from global.
// Lane owns edge column n = lane&15 of its group; it needs concat-feature
// indices [g0..g0+7] and [g0+16..g0+23] (kbase=(lane>>4)*8); each 8-float
// group lies entirely inside one of {x[src], x[dst], edge_attr} (segment
// boundaries 64/128 are multiples of 8, so the select folds after unrolling).
__device__ __forceinline__ const float* seg_ptr(const float* __restrict__ x,
                                                const float* __restrict__ ea,
                                                long sN, long dN, long e, int g) {
    if (g < 64)       return x + sN * 64 + g;
    else if (g < 128) return x + dN * 64 + (g - 64);
    else              return ea + e * 32 + (g - 128);
}

__device__ __forceinline__ v16h make_feat_frag(const float* __restrict__ x,
                                               const float* __restrict__ ea,
                                               long sN, long dN, long e,
                                               int c, int lane) {
    int kbase = (lane >> 4) * 8;
    int g0 = c * 32 + kbase;
    int g1 = g0 + 16;
    const float* p0 = seg_ptr(x, ea, sN, dN, e, g0);
    const float* p1 = seg_ptr(x, ea, sN, dN, e, g1);
    v4f a0 = *(const v4f*)p0;
    v4f a1 = *(const v4f*)(p0 + 4);
    v4f b0 = *(const v4f*)p1;
    v4f b1 = *(const v4f*)(p1 + 4);
    v16h r;
#pragma unroll
    for (int i = 0; i < 4; ++i) {
        r[i]      = (half_t)a0[i];
        r[4 + i]  = (half_t)a1[i];
        r[8 + i]  = (half_t)b0[i];
        r[12 + i] = (half_t)b1[i];
    }
    return r;
}

// Next-layer B fragment for kchunk c is a pure per-lane permutation of the
// current D tiles: halves 0..7 <- D[t=2c][r], halves 8..15 <- D[t=2c+1][r].
__device__ __forceinline__ v16h build_bfrag(const v8f& lo_t, const v8f& hi_t) {
    v16h b;
#pragma unroll
    for (int j = 0; j < 8; ++j) {
        b[j]     = (half_t)lo_t[j];
        b[8 + j] = (half_t)hi_t[j];
    }
    return b;
}

// bias + leaky-relu in place on transposed activations.
// D layout: lane owns edge column; VGPR r of feature tile t holds output
// feature f = t*16 + (lane>=16 ? 8 : 0) + r.
__device__ __forceinline__ void finish_act(v8f acc[4][2],
                                           const float* __restrict__ bias,
                                           int lane) {
    int mb = (lane >> 4) * 8;
#pragma unroll
    for (int t = 0; t < 4; ++t) {
#pragma unroll
        for (int r = 0; r < 8; ++r) {
            float bv = bias[t * 16 + mb + r];
#pragma unroll
            for (int u = 0; u < 2; ++u) {
                float h = acc[t][u][r] + bv;
                acc[t][u][r] = (h >= 0.f) ? h : NEG_SLOPE_C * h;
            }
        }
    }
}

// One kchunk against all 4 output-feature tiles: preload 4 distinct weight
// fragments (8 independent ds_load_b128), then 8 back-to-back WMMAs -- the
// scheduler can use partial dscnt waits instead of full drains.
__device__ __forceinline__ void mma_kchunk(const half_t* lds_w, int frag_off,
                                           v16h b0, v16h b1, v8f (*acc)[2],
                                           int lane) {
    v16h aw[4];
#pragma unroll
    for (int t = 0; t < 4; ++t)
        aw[t] = load_w_lds(lds_w, frag_off + t * FRAG_HALVES, lane);
#pragma unroll
    for (int t = 0; t < 4; ++t) {
        acc[t][0] = __builtin_amdgcn_wmma_f32_16x16x32_f16(
            false, aw[t], false, b0, (short)0, acc[t][0], false, false);
        acc[t][1] = __builtin_amdgcn_wmma_f32_16x16x32_f16(
            false, aw[t], false, b1, (short)0, acc[t][1], false, false);
    }
}

// ---------------------------------------------------------------------------
// Main kernel: 256 threads = 8 waves; each wave processes 32 edges (two
// 16-edge column groups). Whole MLP runs transposed (features x edges), so
// layers chain entirely in registers; LDS holds only the packed weights,
// staged with CDNA5 async global->LDS DMA.
// ---------------------------------------------------------------------------
__global__ void __launch_bounds__(BLOCK_THREADS)
mpconv_kernel(const float* __restrict__ x,
              const int*   __restrict__ ei,       // [2][E]
              const float* __restrict__ ea,
              const float* __restrict__ b_in,
              const float* __restrict__ b_h,      // [4][64]
              const float* __restrict__ b_out,
              const half_t* __restrict__ wpk,
              float* __restrict__ out) {
    __shared__ half_t lds_w[WPK_HALVES];          // 61 440 B
    const int lane = threadIdx.x & 31;

    // ---- stage packed weights into LDS via async DMA (no VGPR round trip;
    //      overlaps with index loads / address setup below) ----
    {
        // low 32 bits of the generic LDS address are the LDS byte offset
        unsigned lds_base = (unsigned)(size_t)(void*)&lds_w[0];
        unsigned long long gbase = (unsigned long long)(const void*)wpk;
#pragma unroll 4
        for (int i = threadIdx.x; i < WPK_HALVES / 8; i += BLOCK_THREADS) {
            unsigned           ldst = lds_base + (unsigned)i * 16u;
            unsigned long long ga   = gbase + (unsigned long long)i * 16u;
            asm volatile("global_load_async_to_lds_b128 %0, %1, off"
                         :: "v"(ldst), "v"(ga) : "memory");
        }
    }

    const int  wave = threadIdx.x >> 5;
    const long base = ((long)blockIdx.x * WAVES_PER_BLOCK + wave) * EDGES_PER_WAVE;
    const int  m    = lane & 15;

    const bool ok0 = (base + m)      < N_EDGES_C;
    const bool ok1 = (base + 16 + m) < N_EDGES_C;
    long e0 = base + m;        if (!ok0) e0 = 0;
    long e1 = base + 16 + m;   if (!ok1) e1 = 0;
    const long s0 = ei[e0], d0 = ei[(long)N_EDGES_C + e0];
    const long s1 = ei[e1], d1 = ei[(long)N_EDGES_C + e1];

    // drain our async copies, then block-wide barrier so all waves see weights
    asm volatile("s_wait_asynccnt 0x0" ::: "memory");
    __syncthreads();

    // acc[t][u]: D tile for output-feature tile t, edge group u (transposed)
    v8f acc[4][2];
#pragma unroll
    for (int t = 0; t < 4; ++t)
#pragma unroll
        for (int u = 0; u < 2; ++u) acc[t][u] = vzero8();

    // ---------------- layer 0: 160 -> 64 ----------------
#pragma unroll
    for (int c = 0; c < 5; ++c) {
        v16h bx0 = make_feat_frag(x, ea, s0, d0, e0, c, lane);
        v16h bx1 = make_feat_frag(x, ea, s1, d1, e1, c, lane);
        mma_kchunk(lds_w, L0_OFF + c * 4 * FRAG_HALVES, bx0, bx1, acc, lane);
    }
    finish_act(acc, b_in, lane);

    // ---------------- 4 hidden layers: 64 -> 64 ----------------
#pragma unroll
    for (int L = 0; L < 4; ++L) {
        v16h bf[2][2];                         // [u][kchunk]
#pragma unroll
        for (int u = 0; u < 2; ++u) {
            bf[u][0] = build_bfrag(acc[0][u], acc[1][u]);
            bf[u][1] = build_bfrag(acc[2][u], acc[3][u]);
        }
        v8f acc2[4][2];
#pragma unroll
        for (int t = 0; t < 4; ++t)
#pragma unroll
            for (int u = 0; u < 2; ++u) acc2[t][u] = vzero8();

#pragma unroll
        for (int c = 0; c < 2; ++c)
            mma_kchunk(lds_w,
                       HID_OFF + (L * HID_FRAGS + c * 4) * FRAG_HALVES,
                       bf[0][c], bf[1][c], acc2, lane);

#pragma unroll
        for (int t = 0; t < 4; ++t)
#pragma unroll
            for (int u = 0; u < 2; ++u) acc[t][u] = acc2[t][u];
        finish_act(acc, b_h + L * OUT_CH_C, lane);
    }

    // ---------------- output layer: 64 -> 64 (linear) ----------------
    {
        v16h bf[2][2];
#pragma unroll
        for (int u = 0; u < 2; ++u) {
            bf[u][0] = build_bfrag(acc[0][u], acc[1][u]);
            bf[u][1] = build_bfrag(acc[2][u], acc[3][u]);
        }
#pragma unroll
        for (int t = 0; t < 4; ++t)
#pragma unroll
            for (int u = 0; u < 2; ++u) acc[t][u] = vzero8();

#pragma unroll
        for (int c = 0; c < 2; ++c)
            mma_kchunk(lds_w, OUT_OFF + c * 4 * FRAG_HALVES,
                       bf[0][c], bf[1][c], acc, lane);
    }

    // ---------------- scatter-sum: lane owns half the features of 2 edge columns ----------
    const int mb = (lane >> 4) * 8;
#pragma unroll
    for (int t = 0; t < 4; ++t) {
#pragma unroll
        for (int r = 0; r < 8; ++r) {
            int   f  = t * 16 + mb + r;
            float bo = b_out[f];
            if (ok0)
                unsafeAtomicAdd(out + d0 * OUT_CH_C + f, acc[t][0][r] + bo);
            if (ok1)
                unsafeAtomicAdd(out + d1 * OUT_CH_C + f, acc[t][1][r] + bo);
        }
    }
}

// ---------------------------------------------------------------------------
extern "C" void kernel_launch(void* const* d_in, const int* in_sizes, int n_in,
                              void* d_out, int out_size, void* d_ws, size_t ws_size,
                              hipStream_t stream) {
    const float* x     = (const float*)d_in[0];
    const int*   ei    = (const int*)d_in[1];
    const float* ea    = (const float*)d_in[2];
    const float* W_in  = (const float*)d_in[3];
    const float* b_in  = (const float*)d_in[4];
    const float* W_h   = (const float*)d_in[5];
    const float* b_h   = (const float*)d_in[6];
    const float* W_out = (const float*)d_in[7];
    const float* b_out = (const float*)d_in[8];
    float*  out = (float*)d_out;
    half_t* wpk = (half_t*)d_ws;

    // 1) pack weights -> f16 fragment layout in scratch (61.4 KB)
    pack_weights_kernel<<<(WPK_HALVES + 255) / 256, 256, 0, stream>>>(
        W_in, W_h, W_out, wpk);

    // 2) zero the output (harness poisons it; we scatter with atomics)
    zero_kernel<<<(out_size + 255) / 256, 256, 0, stream>>>(out, out_size);

    // 3) main fused gather -> transposed-MLP (WMMA) -> scatter kernel
    int nblocks = (N_EDGES_C + EDGES_PER_BLOCK - 1) / EDGES_PER_BLOCK; // 6250
    mpconv_kernel<<<nblocks, BLOCK_THREADS, 0, stream>>>(
        x, ei, ea, b_in, b_h, b_out, wpk, out);
}
